// self_attention_23510650978957
// MI455X (gfx1250) — matrix operand order repaired and checked by reading, not compiled, not run on hardware
//
#include <hip/hip_runtime.h>

// ---------------- problem constants ----------------
namespace {
constexpr int BN = 4;            // batch
constexpr int CH = 256;          // channels
constexpr int C8 = 32;           // CH/8
constexpr int NN = 4096;         // H*W
constexpr int NT = NN / 16;      // 256 tiles along n/j/m
constexpr int MROWS = C8 + C8 + CH; // 320 stacked output rows of GEMM1
constexpr int MT = MROWS / 16;      // 20
}

// ---------------- WMMA types / helpers ----------------
typedef __attribute__((ext_vector_type(16))) __bf16          v16bf;
typedef __attribute__((ext_vector_type(8)))  float           v8f;
typedef __attribute__((ext_vector_type(8)))  unsigned short  ush8;

__device__ __forceinline__ unsigned short f2bf(float f) {
  union { float f; unsigned u; } c; c.f = f;
  unsigned u = c.u;
  u += 0x7fffu + ((u >> 16) & 1u);        // round to nearest even
  return (unsigned short)(u >> 16);
}

// A-matrix 16x32 bf16 tile, per-lane: row = base+(lane&15),
// K = 16*hl + slot (contiguous)
__device__ __forceinline__ v16bf ldA(const unsigned short* p) {
  union { v16bf v; ush8 h[2]; } u;
  u.h[0] = *(const ush8*)p;
  u.h[1] = *(const ush8*)(p + 8);
  return u.v;
}
// B-matrix 32x16 bf16 tile, per-lane: col = lane&15, K = 16*hl + slot (contiguous)
__device__ __forceinline__ v16bf ldB(const unsigned short* p) {
  return *(const v16bf*)p;
}
__device__ __forceinline__ v8f wmma_bf16(v16bf a, v16bf b, v8f c) {
  return __builtin_amdgcn_wmma_f32_16x16x32_bf16(false, a, false, b, (short)0, c,
                                                 false, false);
}
__device__ __forceinline__ v8f vzero8() {
  v8f z = {0.f, 0.f, 0.f, 0.f, 0.f, 0.f, 0.f, 0.f};
  return z;
}
// pack 8 f32 (+bias) into 8 bf16 and store as one 16B write
__device__ __forceinline__ void st_pack8(unsigned short* p, v8f acc,
                                         const float* __restrict__ bias) {
  union { ush8 v; unsigned u[4]; } o;
  #pragma unroll
  for (int i = 0; i < 4; ++i)
    o.u[i] = (unsigned)f2bf(acc[2 * i] + bias[2 * i]) |
             ((unsigned)f2bf(acc[2 * i + 1] + bias[2 * i + 1]) << 16);
  *(ush8*)p = o.v;
}

// ---------------- K1a: x[b][c][n] f32 -> xbT[b][n][c] bf16 ----------------
__global__ void __launch_bounds__(256) k_pack_x(const float* __restrict__ x,
                                                unsigned short* __restrict__ xbT) {
  int idx = blockIdx.x * 256 + threadIdx.x;
  if (idx >= BN * CH * NN) return;
  int n = idx % NN;
  int c = (idx / NN) % CH;
  int b = idx / (NN * CH);
  float v = __builtin_nontemporal_load(&x[idx]);   // streamed once (TH=NT)
  xbT[(size_t)b * NN * CH + (size_t)n * CH + c] = f2bf(v);
}

// ---------------- K1b: weights -> bf16 (Wcat = [W1;W2;W3], W4b), bcat ------
__global__ void __launch_bounds__(256) k_pack_w(
    const float* __restrict__ W1, const float* __restrict__ b1,
    const float* __restrict__ W2, const float* __restrict__ b2,
    const float* __restrict__ W3, const float* __restrict__ b3,
    const float* __restrict__ W4,
    unsigned short* __restrict__ Wcat, float* __restrict__ bcat,
    unsigned short* __restrict__ W4b) {
  int idx = blockIdx.x * 256 + threadIdx.x;
  int total = MROWS * CH + CH * CH;
  if (idx >= total) return;
  if (idx < MROWS * CH) {
    int c = idx % CH;
    int o = idx / CH;
    float v;
    if (o < C8)            v = W1[o * CH + c];
    else if (o < 2 * C8)   v = W2[(o - C8) * CH + c];
    else                   v = W3[(o - 2 * C8) * CH + c];
    Wcat[idx] = f2bf(v);
    if (c == 0)
      bcat[o] = (o < C8) ? b1[o] : (o < 2 * C8) ? b2[o - C8] : b3[o - 2 * C8];
  } else {
    int j = idx - MROWS * CH;
    W4b[j] = f2bf(W4[j]);
  }
}

// ---------------- K2: [320x256] x [256x4096] -> fT / gT / h -------------
// 2x2 register blocking: 4 WMMAs per (A0,A1,B0,B1) load quartet.
// Output routing is WAVE-UNIFORM on mp: mp==0 -> fT, mp==1 -> gT, mp>=2 -> h.
__global__ void __launch_bounds__(256) k_gemm1(
    const unsigned short* __restrict__ Wcat, const float* __restrict__ bcat,
    const unsigned short* __restrict__ xbT,
    unsigned short* __restrict__ fT, unsigned short* __restrict__ gT,
    unsigned short* __restrict__ hM) {
  int w    = blockIdx.x * 8 + (threadIdx.x >> 5);
  int lane = threadIdx.x & 31;
  int hl   = lane >> 4, l16 = lane & 15;
  constexpr int MP = MT / 2, NP = NT / 2;
  int b  = w / (MP * NP);
  int r0 = w % (MP * NP);
  int mp = r0 / NP, np = r0 % NP;

  const unsigned short* arow0 = Wcat + (size_t)(mp * 32 + l16) * CH + 16 * hl;
  const unsigned short* arow1 = arow0 + (size_t)16 * CH;
  const unsigned short* brow0 =
      xbT + (size_t)b * NN * CH + (size_t)(np * 32 + l16) * CH + 16 * hl;
  const unsigned short* brow1 = brow0 + (size_t)16 * CH;

  v8f a00 = vzero8(), a01 = vzero8(), a10 = vzero8(), a11 = vzero8();
  #pragma unroll 4
  for (int kc = 0; kc < CH / 32; ++kc) {
    v16bf wa0 = ldA(arow0 + kc * 32);
    v16bf wa1 = ldA(arow1 + kc * 32);
    v16bf xb0 = ldB(brow0 + kc * 32);
    v16bf xb1 = ldB(brow1 + kc * 32);
    a00 = wmma_bf16(wa0, xb0, a00);
    a01 = wmma_bf16(wa0, xb1, a01);
    a10 = wmma_bf16(wa1, xb0, a10);
    a11 = wmma_bf16(wa1, xb1, a11);
  }

  int n0 = np * 32 + l16, n1 = n0 + 16;
  int rb0 = mp * 32 + hl * 8, rb1 = rb0 + 16;   // 8-row bases of the 2 M-tiles
  float bias0[8], bias1[8];
  #pragma unroll
  for (int r = 0; r < 8; ++r) { bias0[r] = bcat[rb0 + r]; bias1[r] = bcat[rb1 + r]; }

  if (mp == 0) {            // rows 0..31  -> fT[n][row], lane-contiguous in row
    unsigned short* f0 = fT + (size_t)b * NN * C8;
    st_pack8(f0 + (size_t)n0 * C8 + rb0, a00, bias0);
    st_pack8(f0 + (size_t)n1 * C8 + rb0, a01, bias0);
    st_pack8(f0 + (size_t)n0 * C8 + rb1, a10, bias1);
    st_pack8(f0 + (size_t)n1 * C8 + rb1, a11, bias1);
  } else if (mp == 1) {     // rows 32..63 -> gT[n][row-32]
    unsigned short* g0 = gT + (size_t)b * NN * C8;
    st_pack8(g0 + (size_t)n0 * C8 + (rb0 - 32), a00, bias0);
    st_pack8(g0 + (size_t)n1 * C8 + (rb0 - 32), a01, bias0);
    st_pack8(g0 + (size_t)n0 * C8 + (rb1 - 32), a10, bias1);
    st_pack8(g0 + (size_t)n1 * C8 + (rb1 - 32), a11, bias1);
  } else {                  // rows 64..319 -> h[row-64][n], coalesced per r
    unsigned short* h0 = hM + (size_t)b * CH * NN;
    #pragma unroll
    for (int r = 0; r < 8; ++r) {
      size_t ra = (size_t)(rb0 - 64 + r) * NN, rb = (size_t)(rb1 - 64 + r) * NN;
      h0[ra + n0] = f2bf(a00[r] + bias0[r]);
      h0[ra + n1] = f2bf(a01[r] + bias0[r]);
      h0[rb + n0] = f2bf(a10[r] + bias1[r]);
      h0[rb + n1] = f2bf(a11[r] + bias1[r]);
    }
  }
}

// ---------------- K3: per-row softmax adjust  sadj = max + log(sumexp) ----
__global__ void __launch_bounds__(256) k_stats(
    const unsigned short* __restrict__ fT, const unsigned short* __restrict__ gT,
    float* __restrict__ sadj) {
  int w    = blockIdx.x * 8 + (threadIdx.x >> 5);
  int lane = threadIdx.x & 31;
  int hl   = lane >> 4, l16 = lane & 15;
  int b  = w / NT;
  int mt = w % NT;
  const unsigned short* fTb = fT + (size_t)b * NN * C8;
  const unsigned short* gTb = gT + (size_t)b * NN * C8;

  v16bf aT = ldA(fTb + (size_t)(mt * 16 + l16) * C8 + 16 * hl); // K=32 == C8
  v8f z = vzero8();

  float vmax[8];
  #pragma unroll
  for (int r = 0; r < 8; ++r) vmax[r] = -3.0e38f;
  for (int jt = 0; jt < NT; ++jt) {
    v16bf bT = ldB(gTb + (size_t)(jt * 16 + l16) * C8 + 16 * hl);
    v8f d = wmma_bf16(aT, bT, z);
    #pragma unroll
    for (int r = 0; r < 8; ++r) vmax[r] = fmaxf(vmax[r], d[r]);
  }
  #pragma unroll
  for (int r = 0; r < 8; ++r)
    for (int m = 1; m < 16; m <<= 1)           // reduce within 16-lane half
      vmax[r] = fmaxf(vmax[r], __shfl_xor(vmax[r], m, 32));

  float vsum[8];
  #pragma unroll
  for (int r = 0; r < 8; ++r) vsum[r] = 0.f;
  for (int jt = 0; jt < NT; ++jt) {
    v16bf bT = ldB(gTb + (size_t)(jt * 16 + l16) * C8 + 16 * hl);
    v8f d = wmma_bf16(aT, bT, z);
    #pragma unroll
    for (int r = 0; r < 8; ++r) vsum[r] += __expf(d[r] - vmax[r]);
  }
  #pragma unroll
  for (int r = 0; r < 8; ++r)
    for (int m = 1; m < 16; m <<= 1)
      vsum[r] += __shfl_xor(vsum[r], m, 32);
  if (l16 == 0)
    #pragma unroll
    for (int r = 0; r < 8; ++r)   // exp(d - sadj) == exp(d - max) / sumexp
      sadj[b * NN + mt * 16 + hl * 8 + r] = vmax[r] + __logf(vsum[r]);
}

// exp-normalize two 16x16 score D-tiles into one 32x16 bf16 B-tile.
// D layout already has lane = fixed column j; swap M-halves across the two
// 16-lane halves with one shfl_xor(16) to get the B K-ordering.
__device__ __forceinline__ v16bf build_p(v8f d1, v8f d2,
                                         const float* __restrict__ st, int hl) {
  // st: [0..7] for d1 rows, [8..15] for d2 rows (per this lane's half)
  unsigned pk1[4], pk2[4];
  #pragma unroll
  for (int i = 0; i < 4; ++i) {
    float p0 = __expf(d1[2 * i]     - st[2 * i]);
    float p1 = __expf(d1[2 * i + 1] - st[2 * i + 1]);
    float q0 = __expf(d2[2 * i]     - st[8 + 2 * i]);
    float q1 = __expf(d2[2 * i + 1] - st[8 + 2 * i + 1]);
    pk1[i] = (unsigned)f2bf(p0) | ((unsigned)f2bf(p1) << 16);
    pk2[i] = (unsigned)f2bf(q0) | ((unsigned)f2bf(q1) << 16);
  }
  union { v16bf v; unsigned u[8]; } pb;
  #pragma unroll
  for (int i = 0; i < 4; ++i) {
    unsigned sw1 = __shfl_xor(pk1[i], 16, 32);
    unsigned sw2 = __shfl_xor(pk2[i], 16, 32);
    pb.u[i]     = hl ? sw2 : pk1[i];   // K slots 0..7
    pb.u[4 + i] = hl ? pk2[i] : sw1;   // K slots 8..15
  }
  return pb.v;
}

// ---------------- K4: fused  o[:,j] = h @ softmax(f^T g)  ----------------
// wave = (batch, j-tile PAIR, channel-half). fT score A-tiles, h A-tiles and
// softmax stats are shared across the j-tile pair: ~20 WMMAs per ~20 b128.
__global__ void __launch_bounds__(256) k_attn(
    const unsigned short* __restrict__ fT, const unsigned short* __restrict__ gT,
    const unsigned short* __restrict__ hM, const float* __restrict__ sadj,
    unsigned short* __restrict__ oT) {
  int w    = blockIdx.x * 8 + (threadIdx.x >> 5);
  int lane = threadIdx.x & 31;
  int hl   = lane >> 4, l16 = lane & 15;
  int b  = w / NT;           // NT waves per batch = 128 j-pairs * 2 c-halves
  int r0 = w % NT;
  int jp = r0 >> 1;          // j-tile pair 0..127
  int ch = r0 & 1;
  int jt0 = jp * 2, jt1 = jp * 2 + 1;

  const unsigned short* fTb = fT + (size_t)b * NN * C8;
  const unsigned short* gTb = gT + (size_t)b * NN * C8;
  const unsigned short* hMb = hM + (size_t)b * CH * NN;
  const float* sab = sadj + b * NN;

  v16bf gB0 = ldB(gTb + (size_t)(jt0 * 16 + l16) * C8 + 16 * hl); // loop-invariant
  v16bf gB1 = ldB(gTb + (size_t)(jt1 * 16 + l16) * C8 + 16 * hl);
  v8f z = vzero8();
  v8f acc0[8], acc1[8];
  #pragma unroll
  for (int t = 0; t < 8; ++t) { acc0[t] = vzero8(); acc1[t] = vzero8(); }

  for (int mc = 0; mc < NN / 32; ++mc) {
    int m0 = mc * 32;
    v16bf a1 = ldA(fTb + (size_t)(m0 + l16) * C8 + 16 * hl);
    v16bf a2 = ldA(fTb + (size_t)(m0 + 16 + l16) * C8 + 16 * hl);
    if (mc + 1 < NN / 32)   // global_prefetch for the next chunk's fT rows
      __builtin_prefetch(fTb + (size_t)(m0 + 32 + l16) * C8, 0, 3);

    // fused softmax stats (max + log sum), shared by both j-tiles of the pair
    float st[16];
    #pragma unroll
    for (int r = 0; r < 8; ++r) {
      st[r]     = sab[m0 + hl * 8 + r];
      st[8 + r] = sab[m0 + 16 + hl * 8 + r];
    }

    v8f d1 = wmma_bf16(a1, gB0, z);   // scores rows m0..m0+15, cols jt0
    v8f d2 = wmma_bf16(a2, gB0, z);   // scores rows m0+16..m0+31, cols jt0
    v16bf pb0 = build_p(d1, d2, st, hl);
    v8f e1 = wmma_bf16(a1, gB1, z);   // same rows, cols jt1
    v8f e2 = wmma_bf16(a2, gB1, z);
    v16bf pb1 = build_p(e1, e2, st, hl);

    #pragma unroll
    for (int ct = 0; ct < 8; ++ct) {
      int c0 = ch * 128 + ct * 16;
      v16bf ah = ldA(hMb + (size_t)(c0 + l16) * NN + m0 + 16 * hl);
      acc0[ct] = wmma_bf16(ah, pb0, acc0[ct]);  // h tile reused twice
      acc1[ct] = wmma_bf16(ah, pb1, acc1[ct]);
    }
  }

  // store oT[b][j][c] bf16 (contiguous 16B per lane)
  unsigned short* ob0 = oT + (size_t)b * NN * CH + (size_t)(jt0 * 16 + l16) * CH;
  unsigned short* ob1 = oT + (size_t)b * NN * CH + (size_t)(jt1 * 16 + l16) * CH;
  float zb[8];
  #pragma unroll
  for (int r = 0; r < 8; ++r) zb[r] = 0.f;
  #pragma unroll
  for (int ct = 0; ct < 8; ++ct) {
    int c0 = ch * 128 + ct * 16 + 8 * hl;
    st_pack8(ob0 + c0, acc0[ct], zb);
    st_pack8(ob1 + c0, acc1[ct], zb);
  }
}

// ---------------- K5: out = gamma*(W4 @ o + b4) + x ----------------------
// 2x2 register blocking like K2.
__global__ void __launch_bounds__(256) k_gemm2(
    const unsigned short* __restrict__ W4b, const float* __restrict__ b4,
    const unsigned short* __restrict__ oT, const float* __restrict__ x,
    const float* __restrict__ gamma, float* __restrict__ out) {
  int w    = blockIdx.x * 8 + (threadIdx.x >> 5);
  int lane = threadIdx.x & 31;
  int hl   = lane >> 4, l16 = lane & 15;
  constexpr int CP = 8, NP = NT / 2;   // 8 c-tile pairs, 128 n-tile pairs
  int b  = w / (CP * NP);
  int r0 = w % (CP * NP);
  int cp = r0 / NP, np = r0 % NP;

  const unsigned short* arow0 = W4b + (size_t)(cp * 32 + l16) * CH + 16 * hl;
  const unsigned short* arow1 = arow0 + (size_t)16 * CH;
  const unsigned short* brow0 =
      oT + (size_t)b * NN * CH + (size_t)(np * 32 + l16) * CH + 16 * hl;
  const unsigned short* brow1 = brow0 + (size_t)16 * CH;

  v8f a00 = vzero8(), a01 = vzero8(), a10 = vzero8(), a11 = vzero8();
  #pragma unroll 4
  for (int kc = 0; kc < CH / 32; ++kc) {
    v16bf wa0 = ldA(arow0 + kc * 32);
    v16bf wa1 = ldA(arow1 + kc * 32);
    v16bf ob0 = ldB(brow0 + kc * 32);
    v16bf ob1 = ldB(brow1 + kc * 32);
    a00 = wmma_bf16(wa0, ob0, a00);
    a01 = wmma_bf16(wa0, ob1, a01);
    a10 = wmma_bf16(wa1, ob0, a10);
    a11 = wmma_bf16(wa1, ob1, a11);
  }

  float g = gamma[0];
  const float* xb   = x   + (size_t)b * CH * NN;
  float*       outb = out + (size_t)b * CH * NN;
  int n0 = np * 32 + l16, n1 = n0 + 16;
  int c0 = cp * 32 + hl * 8, c1 = c0 + 16;
  #pragma unroll
  for (int r = 0; r < 8; ++r) {
    int ca = c0 + r, cb = c1 + r;
    outb[(size_t)ca * NN + n0] = g * (a00[r] + b4[ca]) + xb[(size_t)ca * NN + n0];
    outb[(size_t)ca * NN + n1] = g * (a01[r] + b4[ca]) + xb[(size_t)ca * NN + n1];
    outb[(size_t)cb * NN + n0] = g * (a10[r] + b4[cb]) + xb[(size_t)cb * NN + n0];
    outb[(size_t)cb * NN + n1] = g * (a11[r] + b4[cb]) + xb[(size_t)cb * NN + n1];
  }
}

// ---------------- host launcher ----------------
extern "C" void kernel_launch(void* const* d_in, const int* in_sizes, int n_in,
                              void* d_out, int out_size, void* d_ws, size_t ws_size,
                              hipStream_t stream) {
  (void)in_sizes; (void)n_in; (void)out_size; (void)ws_size;
  const float* x     = (const float*)d_in[0];
  const float* W1    = (const float*)d_in[1];
  const float* b1    = (const float*)d_in[2];
  const float* W2    = (const float*)d_in[3];
  const float* b2    = (const float*)d_in[4];
  const float* W3    = (const float*)d_in[5];
  const float* b3    = (const float*)d_in[6];
  const float* W4    = (const float*)d_in[7];
  const float* b4    = (const float*)d_in[8];
  const float* gamma = (const float*)d_in[9];
  float* out = (float*)d_out;

  char* ws = (char*)d_ws;
  size_t off = 0;
  auto take = [&](size_t bytes) -> char* {
    char* p = ws + off;
    off = (off + bytes + 255) & ~(size_t)255;
    return p;
  };
  unsigned short* xbT  = (unsigned short*)take((size_t)BN * NN * CH * 2);  // 8 MB
  unsigned short* Wcat = (unsigned short*)take((size_t)MROWS * CH * 2);
  float*          bcat = (float*)take((size_t)MROWS * 4);
  unsigned short* W4b  = (unsigned short*)take((size_t)CH * CH * 2);
  unsigned short* fT   = (unsigned short*)take((size_t)BN * NN * C8 * 2);  // 1 MB
  unsigned short* gT   = (unsigned short*)take((size_t)BN * NN * C8 * 2);  // 1 MB
  unsigned short* hM   = (unsigned short*)take((size_t)BN * CH * NN * 2);  // 8 MB
  float*          sadj = (float*)take((size_t)BN * NN * 4);
  unsigned short* oT   = (unsigned short*)take((size_t)BN * NN * CH * 2);  // 8 MB

  k_pack_x<<<(BN * CH * NN + 255) / 256, 256, 0, stream>>>(x, xbT);
  k_pack_w<<<(MROWS * CH + CH * CH + 255) / 256, 256, 0, stream>>>(
      W1, b1, W2, b2, W3, b3, W4, Wcat, bcat, W4b);
  k_gemm1<<<BN * (MT / 2) * (NT / 2) / 8, 256, 0, stream>>>(Wcat, bcat, xbT,
                                                            fT, gT, hM);
  k_stats<<<BN * NT / 8, 256, 0, stream>>>(fT, gT, sadj);
  k_attn<<<BN * NT / 8, 256, 0, stream>>>(fT, gT, hM, sadj, oT);
  k_gemm2<<<BN * 8 * (NT / 2) / 8, 256, 0, stream>>>(W4b, b4, oT, x, gamma, out);
}
